// Model_12945031430904
// MI455X (gfx1250) — matrix operand order repaired and checked by reading, not compiled, not run on hardware
//
#include <hip/hip_runtime.h>
#include <hip/hip_bf16.h>
#include <math.h>

#define T_LEN   4096
#define NBATCH  16
#define LDS_ROWS   260  // 256 T-tile + 4 halo
#define LDS_STRIDE 40   // halves per row: 80 bytes -> every row 16B aligned

typedef __attribute__((ext_vector_type(16))) _Float16 v16h;
typedef __attribute__((ext_vector_type(8)))  _Float16 v8h;
typedef __attribute__((ext_vector_type(8)))  float    v8f;

union HFrag { v16h v; v8h h[2]; };

// ================= implicit-GEMM Conv1d via V_WMMA_F32_16X16X32_F16 ============
// One workgroup: 64 (Cout) x 256 (T) output tile for one batch.
// 8 waves: wave = (wm in 0..1) x (wn in 0..3); each wave owns 32x64 -> 8 WMMA tiles
// (8 independent accumulator chains per tap to hide WMMA + VMEM latency).
// K-dim of the GEMM = cin_pad * ktaps, walked as (ci-chunk of 32) x (tap),
// with double-buffered LDS staging so fills overlap the matrix pipes.
__global__ __launch_bounds__(256)
void conv1d_wmma(const _Float16* __restrict__ xin, int cin_pad,
                 const _Float16* __restrict__ wgt,   // [cout_pad][ktaps][cin_pad] f16
                 const float* __restrict__ bias,
                 int cout, int ktaps, int pad,
                 _Float16* __restrict__ y16, float* __restrict__ y32, int out_cstride)
{
  __shared__ __align__(16) _Float16 lds[2][LDS_ROWS * LDS_STRIDE];
  const int tid  = threadIdx.x;
  const int lane = tid & 31;
  const int wave = tid >> 5;
  const int wm   = wave & 1;             // M half (32 rows)
  const int wn   = wave >> 1;            // N quarter (64 cols)
  const int hi   = lane >> 4;            // 0: lanes 0-15, 1: lanes 16-31
  const int t0   = blockIdx.x * 256;
  const int cob  = blockIdx.y * 64;
  const int b    = blockIdx.z;

  const _Float16* xb = xin + (size_t)b * cin_pad * T_LEN;

  v8f acc[2][4];
#pragma unroll
  for (int i = 0; i < 2; ++i)
#pragma unroll
    for (int j = 0; j < 4; ++j)
#pragma unroll
      for (int e = 0; e < 8; ++e) acc[i][j][e] = 0.0f;

  // stage 32 channels x 260 t-positions, transposed: buf[t_local][ci]
  auto fill = [&](int cc, _Float16* buf) {
    const int ci0 = cc << 5;
    for (int e = tid; e < 32 * LDS_ROWS; e += 256) {
      int ci = e / LDS_ROWS;
      int r  = e - ci * LDS_ROWS;
      int t  = t0 - pad + r;
      _Float16 val = (_Float16)0.0f;
      if (t >= 0 && t < T_LEN) val = xb[(size_t)(ci0 + ci) * T_LEN + t];
      buf[r * LDS_STRIDE + ci] = val;
    }
  };

  const int nchunks = cin_pad >> 5;
  fill(0, lds[0]);
  for (int cc = 0; cc < nchunks; ++cc) {
    __syncthreads();                       // chunk cc staged; prior readers done
    if (cc + 1 < nchunks) fill(cc + 1, lds[(cc + 1) & 1]);
    const _Float16* cur = lds[cc & 1];
    for (int tap = 0; tap < ktaps; ++tap) {
      // A fragments (weights): 16-bit A layout -> lane<16 holds K {0..7,16..23},
      // lane>=16 holds K {8..15,24..31}: two contiguous 16B global loads.
      HFrag afrag[2];
#pragma unroll
      for (int ms = 0; ms < 2; ++ms) {
        int co = cob + wm * 32 + ms * 16 + (lane & 15);
        const _Float16* wp = wgt + ((size_t)co * ktaps + tap) * cin_pad + (cc << 5);
        afrag[ms].h[0] = *(const v8h*)(wp + hi * 8);
        afrag[ms].h[1] = *(const v8h*)(wp + 16 + hi * 8);
      }
      // B fragments (patches): lanes 0-15 -> K {0..7,16..23}, lanes 16-31 -> K {8..15,24..31},
      // two ds_load_b128 from the transposed LDS tile.
      HFrag bfrag[4];
#pragma unroll
      for (int ns = 0; ns < 4; ++ns) {
        int r = wn * 64 + ns * 16 + (lane & 15) + tap;
        const _Float16* lp = &cur[r * LDS_STRIDE + hi * 8];
        bfrag[ns].h[0] = *(const v8h*)(lp);
        bfrag[ns].h[1] = *(const v8h*)(lp + 16);
      }
#pragma unroll
      for (int ms = 0; ms < 2; ++ms)
#pragma unroll
        for (int ns = 0; ns < 4; ++ns)
          acc[ms][ns] = __builtin_amdgcn_wmma_f32_16x16x32_f16(
              false, afrag[ms].v, false, bfrag[ns].v,
              (short)0, acc[ms][ns], false, false);
    }
  }

  // epilogue: C layout VGPR j -> M=j (lanes 0-15) / M=j+8 (lanes 16-31), N=lane&15
#pragma unroll
  for (int ms = 0; ms < 2; ++ms)
#pragma unroll
    for (int ns = 0; ns < 4; ++ns) {
      int t = t0 + wn * 64 + ns * 16 + (lane & 15);
#pragma unroll
      for (int j = 0; j < 8; ++j) {
        int co = cob + wm * 32 + ms * 16 + hi * 8 + j;
        if (co < cout) {
          float v = acc[ms][ns][j] + bias[co];
          size_t o = ((size_t)b * out_cstride + co) * T_LEN + t;
          if (y32) y32[o] = v;
          else     y16[o] = (_Float16)v;
        }
      }
    }
}

// ================= weight repack: f32 [cout][cin][K] -> f16 [cout_pad][K][cin_pad] ==
__global__ __launch_bounds__(256)
void repack_w(const float* __restrict__ w, _Float16* __restrict__ wf,
              int cout, int cin, int K, int cout_pad, int cin_pad)
{
  int idx = blockIdx.x * 256 + threadIdx.x;
  int total = cout_pad * K * cin_pad;
  if (idx >= total) return;
  int ci = idx % cin_pad;
  int k  = (idx / cin_pad) % K;
  int co = idx / (cin_pad * K);
  _Float16 v = (_Float16)0.0f;
  if (co < cout && ci < cin) v = (_Float16)w[((size_t)co * cin + ci) * K + k];
  wf[idx] = v;
}

// ================= x convert: f32 [B][80][T] -> f16 stride 96 (zero-padded) =========
__global__ __launch_bounds__(256)
void cvt_x(const float* __restrict__ x, _Float16* __restrict__ xo)
{
  int idx = blockIdx.x * 256 + threadIdx.x;       // B*96*T
  int t = idx & (T_LEN - 1);
  int c = (idx >> 12) % 96;
  int b = (idx >> 12) / 96;
  _Float16 v = (_Float16)0.0f;
  if (c < 80) v = (_Float16)x[((size_t)b * 80 + c) * T_LEN + t];
  xo[((size_t)b * 96 + c) * T_LEN + t] = v;
}

// ================= LayerNorm(C) + LeakyReLU(0.01) -> f16 =========================
__global__ __launch_bounds__(256)
void ln_leaky(const _Float16* __restrict__ yin, int C,
              const float* __restrict__ g, const float* __restrict__ be,
              _Float16* __restrict__ xout, int out_cstride)
{
  int p = blockIdx.x * 256 + threadIdx.x;         // B*T threads
  int b = p >> 12, t = p & (T_LEN - 1);
  const _Float16* yp = yin + (size_t)b * C * T_LEN + t;
  float s = 0.f, s2 = 0.f;
  for (int c = 0; c < C; ++c) { float v = (float)yp[(size_t)c * T_LEN]; s += v; s2 += v * v; }
  float m  = s / C;
  float rs = rsqrtf(s2 / C - m * m + 1e-5f);
  _Float16* xp = xout + (size_t)b * out_cstride * T_LEN + t;
  for (int c = 0; c < C; ++c) {
    float v = ((float)yp[(size_t)c * T_LEN] - m) * rs * g[c] + be[c];
    v = v > 0.f ? v : 0.01f * v;
    xp[(size_t)c * T_LEN] = (_Float16)v;
  }
}

// ================= LayerNorm(C) + GLU (C -> C/2) -> f16 ==========================
__global__ __launch_bounds__(256)
void ln_glu(const _Float16* __restrict__ yin, int C,
            const float* __restrict__ g, const float* __restrict__ be,
            _Float16* __restrict__ xout, int out_cstride)
{
  int p = blockIdx.x * 256 + threadIdx.x;
  int b = p >> 12, t = p & (T_LEN - 1);
  const _Float16* yp = yin + (size_t)b * C * T_LEN + t;
  float s = 0.f, s2 = 0.f;
  for (int c = 0; c < C; ++c) { float v = (float)yp[(size_t)c * T_LEN]; s += v; s2 += v * v; }
  float m  = s / C;
  float rs = rsqrtf(s2 / C - m * m + 1e-5f);
  int Ch = C >> 1;
  _Float16* xp = xout + (size_t)b * out_cstride * T_LEN + t;
  for (int c = 0; c < Ch; ++c) {
    float a  = ((float)yp[(size_t)c * T_LEN] - m) * rs * g[c] + be[c];
    float bb = ((float)yp[(size_t)(c + Ch) * T_LEN] - m) * rs * g[c + Ch] + be[c + Ch];
    float o  = a * (1.f / (1.f + expf(-bb)));
    xp[(size_t)c * T_LEN] = (_Float16)o;
  }
}

// ================= speaker embedding: normalize + gather -> f16 [B][128] ==========
__global__ void spk_norm(const float* __restrict__ spk, const int* __restrict__ y,
                         _Float16* __restrict__ yb)
{
  int b = blockIdx.x, c = threadIdx.x;           // 16 x 128
  int row = y[b];
  float s = 0.f;
  for (int d = 0; d < 128; ++d) { float v = spk[row * 128 + d]; s += v * v; }
  float inv = 1.f / (sqrtf(s) + 1e-8f);
  yb[b * 128 + c] = (_Float16)(spk[row * 128 + c] * inv);
}

// ================= codebook normalize: f32 [512][64] ==============================
__global__ void cb_norm(const float* __restrict__ cb, float* __restrict__ cbn)
{
  int j = blockIdx.x, d = threadIdx.x;           // 512 x 64
  float s = 0.f;
  for (int k = 0; k < 64; ++k) { float v = cb[j * 64 + k]; s += v * v; }
  float inv = 1.f / (sqrtf(s) + 1e-8f);
  cbn[j * 64 + d] = cb[j * 64 + d] * inv;
}

// ================= VQ: argmax(z . cbn) over 512 codes, write zq f16 ===============
__global__ __launch_bounds__(256)
void vq_assign(const float* __restrict__ z, const float* __restrict__ cbn,
               _Float16* __restrict__ dout, int cstride)
{
  int p = blockIdx.x * 256 + threadIdx.x;        // B*T threads
  int b = p >> 12, t = p & (T_LEN - 1);
  const float* zp = z + (size_t)b * 64 * T_LEN + t;
  float zr[64];
#pragma unroll
  for (int d = 0; d < 64; ++d) zr[d] = zp[(size_t)d * T_LEN];
  int best = 0; float bestv = -3.4e38f;
  for (int j = 0; j < 512; ++j) {
    float acc = 0.f;
#pragma unroll
    for (int d = 0; d < 64; ++d) acc += zr[d] * cbn[j * 64 + d];   // uniform -> scalar loads
    if (acc > bestv) { bestv = acc; best = j; }
  }
  _Float16* op = dout + (size_t)b * cstride * T_LEN + t;
  const float* cr = cbn + best * 64;
  for (int d = 0; d < 64; ++d) op[(size_t)d * T_LEN] = (_Float16)cr[d];
}

// ================= splat speaker channels into a concat buffer ====================
__global__ __launch_bounds__(256)
void fill_spk(const _Float16* __restrict__ yb, _Float16* __restrict__ xout,
              int c0, int cstride)
{
  int idx = blockIdx.x * 256 + threadIdx.x;      // B*128*T
  int t = idx & (T_LEN - 1);
  int c = (idx >> 12) & 127;
  int b = idx >> 19;
  xout[((size_t)b * cstride + c0 + c) * T_LEN + t] = yb[b * 128 + c];
}

// ======================================================================
extern "C" void kernel_launch(void* const* d_in, const int* in_sizes, int n_in,
                              void* d_out, int out_size, void* d_ws, size_t ws_size,
                              hipStream_t stream)
{
  (void)in_sizes; (void)n_in; (void)out_size; (void)ws_size;
  const float* x        = (const float*)d_in[0];
  const int*   y        = (const int*)d_in[1];
  const float* enc_w[4] = {(const float*)d_in[2], (const float*)d_in[3], (const float*)d_in[4], (const float*)d_in[5]};
  const float* enc_b[4] = {(const float*)d_in[6], (const float*)d_in[7], (const float*)d_in[8], (const float*)d_in[9]};
  const float* enc_g[4] = {(const float*)d_in[10], (const float*)d_in[11], (const float*)d_in[12], (const float*)d_in[13]};
  const float* enc_be[4]= {(const float*)d_in[14], (const float*)d_in[15], (const float*)d_in[16], (const float*)d_in[17]};
  const float* mlp_w    = (const float*)d_in[18];
  const float* mlp_b    = (const float*)d_in[19];
  const float* codebook = (const float*)d_in[20];
  const float* spk_emb  = (const float*)d_in[21];
  const float* dec_w[4] = {(const float*)d_in[22], (const float*)d_in[23], (const float*)d_in[24], (const float*)d_in[25]};
  const float* dec_b[4] = {(const float*)d_in[26], (const float*)d_in[27], (const float*)d_in[28], (const float*)d_in[29]};
  const float* dec_g[3] = {(const float*)d_in[30], (const float*)d_in[31], (const float*)d_in[32]};
  const float* dec_be[3]= {(const float*)d_in[33], (const float*)d_in[34], (const float*)d_in[35]};
  float* out = (float*)d_out;

  char* ws = (char*)d_ws;
  size_t off = 0;
  auto take = [&](size_t bytes) { size_t o = off; off += (bytes + 255) & ~(size_t)255; return o; };
  const size_t BT = (size_t)NBATCH * T_LEN;
  _Float16* actA  = (_Float16*)(ws + take(BT * 640 * 2));
  _Float16* actB  = (_Float16*)(ws + take(BT * 640 * 2));
  _Float16* convh = (_Float16*)(ws + take(BT * 1024 * 2));
  float*    zbuf  = (float*)   (ws + take(BT * 64 * 4));
  float*    cbn   = (float*)   (ws + take(512 * 64 * 4));
  _Float16* yb16  = (_Float16*)(ws + take(NBATCH * 128 * 2));

  // conv layer table: {w, bias, cout, cin, K, cout_pad, cin_pad}
  struct LW { const float* w; const float* bias; int cout, cin, K, cout_pad, cin_pad; _Float16* wf; };
  LW L[9] = {
    { enc_w[0], enc_b[0],  512,  80, 5,  512,  96, nullptr },
    { enc_w[1], enc_b[1],  512, 512, 5,  512, 512, nullptr },
    { enc_w[2], enc_b[2],  512, 512, 5,  512, 512, nullptr },
    { enc_w[3], enc_b[3],  512, 512, 5,  512, 512, nullptr },
    { mlp_w,    mlp_b,      64, 512, 1,   64, 512, nullptr },
    { dec_w[0], dec_b[0], 1024, 192, 5, 1024, 192, nullptr },
    { dec_w[1], dec_b[1], 1024, 640, 5, 1024, 640, nullptr },
    { dec_w[2], dec_b[2], 1024, 640, 5, 1024, 640, nullptr },
    { dec_w[3], dec_b[3],   80, 640, 5,  128, 640, nullptr },
  };
  for (int i = 0; i < 9; ++i) {
    size_t halves = (size_t)L[i].cout_pad * L[i].K * L[i].cin_pad;
    L[i].wf = (_Float16*)(ws + take(halves * 2));
    int total = (int)halves;
    repack_w<<<(total + 255) / 256, 256, 0, stream>>>(
        L[i].w, L[i].wf, L[i].cout, L[i].cin, L[i].K, L[i].cout_pad, L[i].cin_pad);
  }

  // small prep kernels
  cvt_x<<<(int)(BT * 96 / 256), 256, 0, stream>>>(x, actA);
  spk_norm<<<NBATCH, 128, 0, stream>>>(spk_emb, y, yb16);
  cb_norm<<<512, 64, 0, stream>>>(codebook, cbn);

  auto conv = [&](const LW& l, const _Float16* in, int pad,
                  _Float16* o16, float* o32, int ostride) {
    dim3 grid(T_LEN / 256, l.cout_pad / 64, NBATCH);
    conv1d_wmma<<<grid, 256, 0, stream>>>(in, l.cin_pad, l.wf, l.bias,
                                          l.cout, l.K, pad, o16, o32, ostride);
  };
  const int POS = (int)(BT / 256);   // blocks for per-(b,t) kernels

  // ---- encoder ----
  conv(L[0], actA, 2, convh, nullptr, 512);
  ln_leaky<<<POS, 256, 0, stream>>>(convh, 512, enc_g[0], enc_be[0], actB, 512);
  conv(L[1], actB, 2, convh, nullptr, 512);
  ln_leaky<<<POS, 256, 0, stream>>>(convh, 512, enc_g[1], enc_be[1], actA, 512);
  conv(L[2], actA, 2, convh, nullptr, 512);
  ln_leaky<<<POS, 256, 0, stream>>>(convh, 512, enc_g[2], enc_be[2], actB, 512);
  conv(L[3], actB, 2, convh, nullptr, 512);
  ln_leaky<<<POS, 256, 0, stream>>>(convh, 512, enc_g[3], enc_be[3], actA, 512);

  // ---- 1x1 conv to z, then VQ ----
  conv(L[4], actA, 0, nullptr, zbuf, 64);
  vq_assign<<<POS, 256, 0, stream>>>(zbuf, cbn, actB, 192);
  fill_spk<<<(int)(BT * 128 / 256), 256, 0, stream>>>(yb16, actB, 64, 192);

  // ---- decoder ----
  conv(L[5], actB, 2, convh, nullptr, 1024);
  ln_glu<<<POS, 256, 0, stream>>>(convh, 1024, dec_g[0], dec_be[0], actA, 640);
  fill_spk<<<(int)(BT * 128 / 256), 256, 0, stream>>>(yb16, actA, 512, 640);

  conv(L[6], actA, 2, convh, nullptr, 1024);
  ln_glu<<<POS, 256, 0, stream>>>(convh, 1024, dec_g[1], dec_be[1], actB, 640);
  fill_spk<<<(int)(BT * 128 / 256), 256, 0, stream>>>(yb16, actB, 512, 640);

  conv(L[7], actB, 2, convh, nullptr, 1024);
  ln_glu<<<POS, 256, 0, stream>>>(convh, 1024, dec_g[2], dec_be[2], actA, 640);
  fill_spk<<<(int)(BT * 128 / 256), 256, 0, stream>>>(yb16, actA, 512, 640);

  // ---- final conv straight to f32 output ----
  conv(L[8], actA, 2, nullptr, out, 80);
}
